// BasicAttention_90838558310712
// MI455X (gfx1250) — matrix operand order repaired and checked by reading, not compiled, not run on hardware
//
#include <hip/hip_runtime.h>
#include <hip/hip_bf16.h>
#include <stdint.h>

// ---------------------------------------------------------------------------
// BasicAttention on MI455X (gfx1250, wave32, WMMA).
// B=8, QL=KL=2048, D=1024. ~240 GFLOP vs ~270MB HBM -> strongly compute
// bound -> bf16 v_wmma_f32_16x16x32_bf16 with f32 accumulation everywhere.
//
// All GEMM shapes are compile-time constants (template params) so the C
// epilogue collapses to immediate-offset clause stores instead of u64 VALU
// address chains, and wave tiles are 64x64 (4x4 subtiles, 128 accum VGPRs)
// for 16 wmma per 8 fragment loads.
// ---------------------------------------------------------------------------

typedef __attribute__((ext_vector_type(16))) __bf16 v16bf;
typedef __attribute__((ext_vector_type(8)))  float  v8f;

union FragU {
    v16bf v;
    uint4 q[2];
    unsigned short s[16];
};

__device__ inline unsigned short f2bf(float f) {
    unsigned int u = __float_as_uint(f);
    u += 0x7FFFu + ((u >> 16) & 1u);
    return (unsigned short)(u >> 16);
}

// A fragment, 16x32 bf16 (MxK). Lane L: M=L&15;
// lanes 0-15 hold K=0..7,16..23 ; lanes 16-31 hold K=8..15,24..31.
template <int LD>
__device__ inline v16bf load_frag_a(const unsigned short* tile) {
    const int lane = threadIdx.x & 31;
    const int m  = lane & 15;
    const int kb = (lane >> 4) << 3;
    const unsigned short* p = tile + m * LD + kb;
    FragU f;
    f.q[0] = *(const uint4*)(p);
    f.q[1] = *(const uint4*)(p + 16);
    return f.v;
}

// B fragment for C = A * B^T with B row-major [N x K]: wmma-B[k][n]=mem[n][k].
// Lane L: N=L&15; lanes 0-15 K=0..7,16..23 ; lanes 16-31 K=8..15,24..31.
template <int LD>
__device__ inline v16bf load_frag_bt(const unsigned short* tile) {
    const int lane = threadIdx.x & 31;
    const int n  = lane & 15;
    const int kb = (lane >> 4) << 3;
    const unsigned short* p = tile + n * LD + kb;
    FragU f;
    f.q[0] = *(const uint4*)(p);
    f.q[1] = *(const uint4*)(p + 16);
    return f.v;
}

__device__ inline v8f wmma_bf16(v16bf a, v16bf b, v8f c) {
    return __builtin_amdgcn_wmma_f32_16x16x32_bf16(
        false, a, false, b, (short)0, c, false, false);
}

// ---------------------------------------------------------------------------
// fp32 -> bf16 one-shot conversion (memory bound)
// ---------------------------------------------------------------------------
__global__ __launch_bounds__(256) void cvt_f32_bf16(
    const float* __restrict__ in, unsigned short* __restrict__ out, int n) {
    int i = blockIdx.x * blockDim.x + threadIdx.x;
    const int stride = gridDim.x * blockDim.x;
    for (; i < n; i += stride) out[i] = f2bf(in[i]);
}

// ---------------------------------------------------------------------------
// NT GEMM: C[m,n] = alpha * sum_k A[m,k]*B[n,k] (+bias[n] if BIAS).
// A:[M x K], B:[N x K] bf16 row-major. 256 threads = 8 waves (2x4);
// block tile 128x256, wave tile 64x64 (4x4 subtiles). No LDS: both fragment
// layouts are contiguous 16B runs per lane, streamed from global/L2.
// ---------------------------------------------------------------------------
template <bool OUT_BF16, bool BIAS, int LDA, int LDB, int LDC, int K,
          long long SA, long long SB, long long SC>
__global__ __launch_bounds__(256) void gemm_nt(
    const unsigned short* __restrict__ A, const unsigned short* __restrict__ B,
    void* __restrict__ Cv, const float* __restrict__ bias, float alpha) {
    const int z = blockIdx.z;
    A += z * SA;
    B += z * SB;

    const int wave = threadIdx.x >> 5;
    const int lane = threadIdx.x & 31;
    const int m0 = blockIdx.x * 128 + (wave & 1) * 64;
    const int n0 = blockIdx.y * 256 + (wave >> 1) * 64;

    const v8f zero = {0.f, 0.f, 0.f, 0.f, 0.f, 0.f, 0.f, 0.f};
    v8f acc[4][4];
#pragma unroll
    for (int i = 0; i < 4; ++i)
#pragma unroll
        for (int j = 0; j < 4; ++j) acc[i][j] = zero;

    for (int kc = 0; kc < K; kc += 32) {
        if (kc + 32 < K) {
            __builtin_prefetch(A + m0 * LDA + kc + 32, 0, 3);
            __builtin_prefetch(B + n0 * LDB + kc + 32, 0, 3);
        }
        v16bf a[4], b[4];
#pragma unroll
        for (int i = 0; i < 4; ++i)
            a[i] = load_frag_a<LDA>(A + (m0 + i * 16) * LDA + kc);
#pragma unroll
        for (int j = 0; j < 4; ++j)
            b[j] = load_frag_bt<LDB>(B + (n0 + j * 16) * LDB + kc);
#pragma unroll
        for (int i = 0; i < 4; ++i)
#pragma unroll
            for (int j = 0; j < 4; ++j)
                acc[i][j] = wmma_bf16(a[i], b[j], acc[i][j]);
    }

    // C/D layout: VGPR r -> (M=r, N=lane)        lanes 0-15
    //                       (M=r+8, N=lane-16)   lanes 16-31
    const int ncol = lane & 15;
    const int moff = (lane >> 4) << 3;
#pragma unroll
    for (int i = 0; i < 4; ++i) {
#pragma unroll
        for (int j = 0; j < 4; ++j) {
            const int row0 = m0 + i * 16 + moff;
            const int col  = n0 + j * 16 + ncol;
            float bv = 0.f;
            if (BIAS) bv = bias[col];
#pragma unroll
            for (int r = 0; r < 8; ++r) {
                const float val = acc[i][j][r] * alpha + bv;
                const long long off = z * SC + (long long)(row0 + r) * LDC + col;
                if (OUT_BF16)
                    ((unsigned short*)Cv)[off] = f2bf(val);
                else
                    ((float*)Cv)[off] = val;
            }
        }
    }
}

// ---------------------------------------------------------------------------
// NN GEMM: C[m,n] = sum_k A[m,k]*B[k,n]. A bf16 [M x K] (padded LDA ok),
// B bf16 [K x N] row-major, staged transposed into padded LDS
// (80B row stride -> 16B aligned + bank-conflict-free ds_load_b128).
// ---------------------------------------------------------------------------
template <int LDA, int LDB, int LDC, int K, long long SA, long long SB,
          long long SC>
__global__ __launch_bounds__(256) void gemm_nn(
    const unsigned short* __restrict__ A, const unsigned short* __restrict__ B,
    float* __restrict__ C) {
    __shared__ unsigned short Bt[256][40];  // [n][k]
    const int z = blockIdx.z;
    A += z * SA;
    B += z * SB;
    C += z * SC;

    const int wave = threadIdx.x >> 5;
    const int lane = threadIdx.x & 31;
    const int m0   = blockIdx.x * 128 + (wave & 1) * 64;
    const int nblk = blockIdx.y * 256;
    const int nw   = (wave >> 1) * 64;

    const v8f zero = {0.f, 0.f, 0.f, 0.f, 0.f, 0.f, 0.f, 0.f};
    v8f acc[4][4];
#pragma unroll
    for (int i = 0; i < 4; ++i)
#pragma unroll
        for (int j = 0; j < 4; ++j) acc[i][j] = zero;

    const int kr = threadIdx.x >> 3;        // 0..31 k-row of chunk
    const int nc = (threadIdx.x & 7) * 32;  // 0..224 n-column base

    for (int kc = 0; kc < K; kc += 32) {
        // stage + transpose B chunk [32 x 256] into LDS
        {
            const unsigned short* src = B + (kc + kr) * LDB + nblk + nc;
            FragU u0, u1;
            u0.q[0] = *(const uint4*)(src);
            u0.q[1] = *(const uint4*)(src + 8);
            u1.q[0] = *(const uint4*)(src + 16);
            u1.q[1] = *(const uint4*)(src + 24);
#pragma unroll
            for (int i = 0; i < 16; ++i) Bt[nc + i][kr] = u0.s[i];
#pragma unroll
            for (int i = 0; i < 16; ++i) Bt[nc + 16 + i][kr] = u1.s[i];
        }
        __syncthreads();

        v16bf a[4], b[4];
#pragma unroll
        for (int i = 0; i < 4; ++i)
            a[i] = load_frag_a<LDA>(A + (m0 + i * 16) * LDA + kc);
#pragma unroll
        for (int j = 0; j < 4; ++j)
            b[j] = load_frag_bt<40>(&Bt[nw + j * 16][0]);
#pragma unroll
        for (int i = 0; i < 4; ++i)
#pragma unroll
            for (int j = 0; j < 4; ++j)
                acc[i][j] = wmma_bf16(a[i], b[j], acc[i][j]);
        __syncthreads();
    }

    const int ncol = lane & 15;
    const int moff = (lane >> 4) << 3;
#pragma unroll
    for (int i = 0; i < 4; ++i) {
#pragma unroll
        for (int j = 0; j < 4; ++j) {
            const int row0 = m0 + i * 16 + moff;
            const int col  = nblk + nw + j * 16 + ncol;
#pragma unroll
            for (int r = 0; r < 8; ++r)
                C[(long long)(row0 + r) * LDC + col] = acc[i][j][r];
        }
    }
}

// ---------------------------------------------------------------------------
// Row softmax over 2048 f32 scores; writes bf16 probabilities in place at
// bf16 row stride 4096 (first half of this row's own f32 bytes -> no
// cross-block aliasing).
// ---------------------------------------------------------------------------
__global__ __launch_bounds__(256) void softmax_row(float* __restrict__ S) {
    __shared__ float red[256];
    const long long row = blockIdx.x;
    const float* rp = S + row * 2048;
    unsigned short* op = (unsigned short*)S + row * 4096;
    const int t = threadIdx.x;

    float4 v0 = *(const float4*)(rp + t * 8);
    float4 v1 = *(const float4*)(rp + t * 8 + 4);
    float vals[8] = {v0.x, v0.y, v0.z, v0.w, v1.x, v1.y, v1.z, v1.w};

    float mx = vals[0];
#pragma unroll
    for (int i = 1; i < 8; ++i) mx = fmaxf(mx, vals[i]);
    red[t] = mx;
    __syncthreads();
    for (int s = 128; s > 0; s >>= 1) {
        if (t < s) red[t] = fmaxf(red[t], red[t + s]);
        __syncthreads();
    }
    mx = red[0];
    __syncthreads();

    float e[8];
    float sum = 0.f;
#pragma unroll
    for (int i = 0; i < 8; ++i) {
        e[i] = __expf(vals[i] - mx);
        sum += e[i];
    }
    red[t] = sum;
    __syncthreads();
    for (int s = 128; s > 0; s >>= 1) {
        if (t < s) red[t] += red[t + s];
        __syncthreads();
    }
    const float inv = 1.f / red[0];
#pragma unroll
    for (int i = 0; i < 8; ++i) op[t * 8 + i] = f2bf(e[i] * inv);
}

// ---------------------------------------------------------------------------
// Host-side orchestration
// ---------------------------------------------------------------------------
extern "C" void kernel_launch(void* const* d_in, const int* in_sizes, int n_in,
                              void* d_out, int out_size, void* d_ws,
                              size_t ws_size, hipStream_t stream) {
    (void)in_sizes; (void)n_in; (void)out_size; (void)ws_size;

    const float* q_embd = (const float*)d_in[0];
    const float* k_embd = (const float*)d_in[1];
    const float* v_embd = (const float*)d_in[2];
    const float* Wq = (const float*)d_in[3];
    const float* bq = (const float*)d_in[4];
    const float* Wk = (const float*)d_in[5];
    const float* bk = (const float*)d_in[6];
    const float* Wv = (const float*)d_in[7];
    const float* bv = (const float*)d_in[8];

    constexpr long long L = 2048, D = 1024;
    constexpr long long M = 8 * L;              // 16384
    const size_t SZ_X = (size_t)M * D * 2;      // 33.5 MB bf16 [M x D]
    const size_t SZ_W = (size_t)D * D * 2;      // 2 MB

    char* w = (char*)d_ws;
    unsigned short* Xq  = (unsigned short*)(w);
    unsigned short* Xk  = (unsigned short*)(w + 1 * SZ_X);
    unsigned short* Xv  = (unsigned short*)(w + 2 * SZ_X);
    unsigned short* Wqb = (unsigned short*)(w + 3 * SZ_X);
    unsigned short* Wkb = (unsigned short*)(w + 3 * SZ_X + 1 * SZ_W);
    unsigned short* Wvb = (unsigned short*)(w + 3 * SZ_X + 2 * SZ_W);
    unsigned short* Qb  = (unsigned short*)(w + 3 * SZ_X + 3 * SZ_W);
    unsigned short* Kb  = (unsigned short*)(w + 4 * SZ_X + 3 * SZ_W);
    unsigned short* Vb  = (unsigned short*)(w + 5 * SZ_X + 3 * SZ_W);
    float* Sf = (float*)(w + 6 * SZ_X + 3 * SZ_W);  // 8*2048*2048 f32

    const dim3 blk(256);

    // 1) fp32 -> bf16
    cvt_f32_bf16<<<2048, blk, 0, stream>>>(q_embd, Xq, (int)(M * D));
    cvt_f32_bf16<<<2048, blk, 0, stream>>>(k_embd, Xk, (int)(M * D));
    cvt_f32_bf16<<<2048, blk, 0, stream>>>(v_embd, Xv, (int)(M * D));
    cvt_f32_bf16<<<512, blk, 0, stream>>>(Wq, Wqb, (int)(D * D));
    cvt_f32_bf16<<<512, blk, 0, stream>>>(Wk, Wkb, (int)(D * D));
    cvt_f32_bf16<<<512, blk, 0, stream>>>(Wv, Wvb, (int)(D * D));

    // 2) projections: Q/K/V = X @ W^T + b   (M=16384, N=1024, K=1024)
    const dim3 gproj(128, 4, 1);
    gemm_nt<true, true, 1024, 1024, 1024, 1024, 0, 0, 0>
        <<<gproj, blk, 0, stream>>>(Xq, Wqb, Qb, bq, 1.0f);
    gemm_nt<true, true, 1024, 1024, 1024, 1024, 0, 0, 0>
        <<<gproj, blk, 0, stream>>>(Xk, Wkb, Kb, bk, 1.0f);
    gemm_nt<true, true, 1024, 1024, 1024, 1024, 0, 0, 0>
        <<<gproj, blk, 0, stream>>>(Xv, Wvb, Vb, bv, 1.0f);

    // 3) scores: S = (Q @ K^T) / sqrt(D)  per batch (2048x2048, K=1024)
    const dim3 gsc(16, 8, 8);
    gemm_nt<false, false, 1024, 1024, 2048, 1024, L * D, L * D, L * L>
        <<<gsc, blk, 0, stream>>>(Qb, Kb, Sf, nullptr, 0.03125f);

    // 4) softmax rows -> bf16 P in place (bf16 row stride 4096)
    softmax_row<<<(int)M, blk, 0, stream>>>(Sf);

    // 5) out = P @ V  per batch (2048x1024, K=2048), f32 result
    const dim3 gnn(16, 4, 8);
    gemm_nn<4096, 1024, 1024, 2048, L * 4096, L * D, L * D>
        <<<gnn, blk, 0, stream>>>((const unsigned short*)Sf, Vb,
                                  (float*)d_out);
}
